// DeltaOnlyModel_3204045603543
// MI455X (gfx1250) — matrix operand order repaired and checked by reading, not compiled, not run on hardware
//
#include <hip/hip_runtime.h>

typedef __bf16 bf16;
typedef bf16  v16bf __attribute__((ext_vector_type(16)));
typedef bf16  v8bf  __attribute__((ext_vector_type(8)));
typedef float v8f   __attribute__((ext_vector_type(8)));

#define H     64
#define H2    128
#define VOC   64
#define SEQL  2048
#define CH    16
#define NCHUNK (SEQL / CH)

// ---------------------------------------------------------------------------
// WMMA operand loaders (wave32; A/B use this code's contiguous k order, the same in both, not AMD's table order)
// A (16x32 bf16): lane L holds row M=L%16; element e -> K = kb + (L/16)*16 + e.
// B (32x16 bf16): lane L holds col N=L%16; element e -> K = kb + (L/16)*16 + e.
// C/D (16x16 f32): element r -> row M = r + (L/16)*8, col N = L%16.
// ---------------------------------------------------------------------------
__device__ __forceinline__ v16bf load_a_tile(const bf16* sm, int stride, int kbase, int lane) {
  const bf16* p = sm + (lane & 15) * stride + kbase + ((lane >> 4) * 16);
  v8bf lo = *(const v8bf*)p;
  v8bf hi = *(const v8bf*)(p + 8);
  v16bf a;
#pragma unroll
  for (int e = 0; e < 8; ++e) { a[e] = lo[e]; a[e + 8] = hi[e]; }
  return a;
}

// wT is the transposed bf16 weight [N][K] row-major -> contiguous 32B per lane.
__device__ __forceinline__ v16bf load_b_tile(const bf16* wT, int kdim, int nbase, int kbase, int lane) {
  return *(const v16bf*)(wT + (nbase + (lane & 15)) * kdim + kbase + ((lane >> 4) * 16));
}

// ---------------------------------------------------------------------------
// Prep: transpose + convert GEMM weights to bf16 into workspace (~48 KB).
// ---------------------------------------------------------------------------
__global__ void prep_weights_kernel(const float* __restrict__ w1,
                                    const float* __restrict__ w2,
                                    const float* __restrict__ wk,
                                    const float* __restrict__ wv,
                                    bf16* __restrict__ ws) {
  int i = blockIdx.x * blockDim.x + threadIdx.x;
  bf16* w1T = ws;                 // [128][64]
  bf16* w2T = w1T + H2 * H;       // [64][128]
  bf16* wkT = w2T + H * H2;       // [64][64]
  bf16* wvT = wkT + H * H;        // [64][64]
  if (i < H2 * H) {
    int n = i / H, k = i % H;
    w1T[i] = (bf16)w1[k * H2 + n];
  } else if (i < 2 * H2 * H) {
    int j = i - H2 * H;
    int n = j / H2, k = j % H2;
    w2T[j] = (bf16)w2[k * H + n];
  } else if (i < 2 * H2 * H + H * H) {
    int j = i - 2 * H2 * H;
    int n = j / H, k = j % H;
    wkT[j] = (bf16)wk[k * H + n];
  } else if (i < 2 * H2 * H + 2 * H * H) {
    int j = i - 2 * H2 * H - H * H;
    int n = j / H, k = j % H;
    wvT[j] = (bf16)wv[k * H + n];
  }
}

// ---------------------------------------------------------------------------
// Fused encoder + delta-rule scan. One workgroup per batch element.
// 128 threads = 4 wave32s. The fast-weight matrix M (64x64 f32) is
// REGISTER-RESIDENT: thread tid owns M[tid&63][(tid>>6)*32 .. +31] (32 VGPRs).
// ---------------------------------------------------------------------------
__global__ __launch_bounds__(128) void delta_fused_kernel(
    const int*   __restrict__ x,
    const float* __restrict__ embed,
    const float* __restrict__ b1v,
    const float* __restrict__ b2v,
    const float* __restrict__ ln_g,
    const float* __restrict__ ln_b,
    const float* __restrict__ wq,
    const float* __restrict__ wo,
    const float* __restrict__ bo,
    const bf16*  __restrict__ w1T,
    const bf16*  __restrict__ w2T,
    const bf16*  __restrict__ wkT,
    const bf16*  __restrict__ wvT,
    float*       __restrict__ out) {
  __shared__ bf16  s_hbf[CH][H];        // post-LN h tile, bf16 A-operand
  __shared__ bf16  s_ffbf[CH][H2];      // relu(h@w1+b1), bf16 A-operand
  __shared__ float s_h[CH][H];          // h tile f32 (embed, then post-LN)
  __shared__ float s_ff2[CH][H];        // ffn output pre-residual
  __shared__ float s_k[CH][H];          // normalized keys
  __shared__ float s_v[CH][H];          // values
  __shared__ float s_pp[2][H];          // partial dot products (2 column halves)
  __shared__ float s_delta[H];
  __shared__ float s_b1[H2];
  __shared__ float s_b2[H], s_g[H], s_be[H];
  __shared__ float s_red[2][2];
  __shared__ float s_gate;
  __shared__ float s_q[H], s_read[H];

  const int b    = blockIdx.x;
  const int tid  = threadIdx.x;
  const int lane = tid & 31;
  const int wave = tid >> 5;
  const int mi   = tid & 63;            // owned M row
  const int jb   = (tid >> 6) * 32;     // owned M column half base

  // register-resident fast-weight slice M[mi][jb..jb+31]
  float m[32];
#pragma unroll
  for (int jj = 0; jj < 32; ++jj) m[jj] = 0.f;

  if (tid < H2) s_b1[tid] = b1v[tid];
  if (tid < H) { s_b2[tid] = b2v[tid]; s_g[tid] = ln_g[tid]; s_be[tid] = ln_b[tid]; }
  __syncthreads();

  for (int chunk = 0; chunk < NCHUNK; ++chunk) {
    const int t0 = chunk * CH;

    // ---- embed gather: 16 tokens x 64 dims ----
    for (int e = tid; e < CH * H; e += 128) {
      int r = e >> 6, c = e & 63;
      int tok = x[b * SEQL + t0 + r];
      float val = embed[tok * H + c];
      s_h[r][c] = val;
      s_hbf[r][c] = (bf16)val;
    }
    __syncthreads();

    // ---- GEMM1: h[16x64] @ w1[64x128], +b1, relu -> s_ffbf ----
#pragma unroll
    for (int nt = 0; nt < 2; ++nt) {
      const int nbase = (wave + nt * 4) * 16;
      v8f acc = {};
#pragma unroll
      for (int kb = 0; kb < H; kb += 32) {
        v16bf a  = load_a_tile(&s_hbf[0][0], H, kb, lane);
        v16bf bm = load_b_tile(w1T, H, nbase, kb, lane);
        acc = __builtin_amdgcn_wmma_f32_16x16x32_bf16(false, a, false, bm,
                                                      (short)0, acc, false, false);
      }
      const int col  = nbase + (lane & 15);
      const float bias = s_b1[col];
#pragma unroll
      for (int r = 0; r < 8; ++r) {
        float v = acc[r] + bias;
        s_ffbf[r + (lane >> 4) * 8][col] = (bf16)fmaxf(v, 0.f);
      }
    }
    __syncthreads();

    // ---- GEMM2: ff[16x128] @ w2[128x64], +b2 -> s_ff2 ----
    {
      const int nbase = wave * 16;
      v8f acc = {};
#pragma unroll
      for (int kb = 0; kb < H2; kb += 32) {
        v16bf a  = load_a_tile(&s_ffbf[0][0], H2, kb, lane);
        v16bf bm = load_b_tile(w2T, H2, nbase, kb, lane);
        acc = __builtin_amdgcn_wmma_f32_16x16x32_bf16(false, a, false, bm,
                                                      (short)0, acc, false, false);
      }
      const int col = nbase + (lane & 15);
#pragma unroll
      for (int r = 0; r < 8; ++r)
        s_ff2[r + (lane >> 4) * 8][col] = acc[r] + s_b2[col];
    }
    __syncthreads();

    // ---- residual + LayerNorm (f32), refresh s_h / s_hbf ----
    if (tid < CH) {
      float sum = 0.f, ssq = 0.f;
#pragma unroll
      for (int j = 0; j < H; ++j) {
        float p = s_h[tid][j] + s_ff2[tid][j];
        sum += p; ssq += p * p;
      }
      float mu   = sum * (1.f / H);
      float var  = ssq * (1.f / H) - mu * mu;
      float rstd = rsqrtf(var + 1e-5f);
#pragma unroll
      for (int j = 0; j < H; ++j) {
        float p  = s_h[tid][j] + s_ff2[tid][j];
        float hn = (p - mu) * rstd * s_g[j] + s_be[j];
        s_h[tid][j]   = hn;
        s_hbf[tid][j] = (bf16)hn;
      }
    }
    __syncthreads();

    // ---- GEMM3/4: k = h@wk, v = h@wv (8 output tiles, 2 per wave) ----
#pragma unroll
    for (int nt = 0; nt < 2; ++nt) {
      const int mtile = wave * 2 + nt;
      const bf16* wT = (mtile < 4) ? wkT : wvT;
      float* dst     = (mtile < 4) ? &s_k[0][0] : &s_v[0][0];
      const int nbase = (mtile & 3) * 16;
      v8f acc = {};
#pragma unroll
      for (int kb = 0; kb < H; kb += 32) {
        v16bf a  = load_a_tile(&s_hbf[0][0], H, kb, lane);
        v16bf bm = load_b_tile(wT, H, nbase, kb, lane);
        acc = __builtin_amdgcn_wmma_f32_16x16x32_bf16(false, a, false, bm,
                                                      (short)0, acc, false, false);
      }
      const int col = nbase + (lane & 15);
#pragma unroll
      for (int r = 0; r < 8; ++r)
        dst[(r + (lane >> 4) * 8) * H + col] = acc[r];
    }
    __syncthreads();

    // ---- L2-normalize key rows ----
    if (tid < CH) {
      float ss = 0.f;
#pragma unroll
      for (int j = 0; j < H; ++j) ss += s_k[tid][j] * s_k[tid][j];
      float inv = 1.f / fmaxf(sqrtf(ss), 1e-12f);
#pragma unroll
      for (int j = 0; j < H; ++j) s_k[tid][j] *= inv;
    }
    __syncthreads();

    // ---- sequential delta-rule scan over the 16 steps ----
    // M register-resident; per step: 32 reg FMAs (pred) + 32 reg FMAs (update).
    for (int t = 0; t < CH; ++t) {
      // cache this step's k segment in registers (vectorized LDS loads)
      float kr[32];
      {
        const float4* kp = (const float4*)&s_k[t][jb];
#pragma unroll
        for (int q4 = 0; q4 < 8; ++q4) {
          float4 kk = kp[q4];
          kr[q4 * 4 + 0] = kk.x; kr[q4 * 4 + 1] = kk.y;
          kr[q4 * 4 + 2] = kk.z; kr[q4 * 4 + 3] = kk.w;
        }
      }
      // partial pred over owned column half
      float part = 0.f;
#pragma unroll
      for (int jj = 0; jj < 32; ++jj) part = fmaf(m[jj], kr[jj], part);
      s_pp[tid >> 6][mi] = part;
      __syncthreads();

      float dsq = 0.f, vsq = 0.f;
      if (tid < H) {
        float pred = s_pp[0][tid] + s_pp[1][tid];
        float d = s_v[t][tid] - pred;
        s_delta[tid] = d;
        dsq = d * d;
        vsq = s_v[t][tid] * s_v[t][tid];
      }
#pragma unroll
      for (int off = 16; off >= 1; off >>= 1) {
        dsq += __shfl_down(dsq, off);
        vsq += __shfl_down(vsq, off);
      }
      if ((tid == 0) || (tid == 32)) { s_red[wave][0] = dsq; s_red[wave][1] = vsq; }
      __syncthreads();
      if (tid == 0) {
        float dn = s_red[0][0] + s_red[1][0];
        float vn = s_red[0][1] + s_red[1][1];
        // ||delta|| > THR*||v||  <=>  dn > THR^2 * vn   (THR = 0.4)
        s_gate = (dn > 0.16f * vn) ? 1.f : 0.f;
      }
      __syncthreads();

      const float gd = s_gate * s_delta[mi];
#pragma unroll
      for (int jj = 0; jj < 32; ++jj) m[jj] = fmaf(gd, kr[jj], m[jj]);
    }
  }

  // ---- epilogue: q = h_last@wq ; read = relu(M q) ; out = read@wo + bo ----
  if (tid < H) {
    float q = 0.f;
#pragma unroll
    for (int i = 0; i < H; ++i) q = fmaf(s_h[CH - 1][i], wq[i * H + tid], q);
    s_q[tid] = q;
  }
  __syncthreads();
  {
    float part = 0.f;
#pragma unroll
    for (int jj = 0; jj < 32; ++jj) part = fmaf(m[jj], s_q[jb + jj], part);
    s_pp[tid >> 6][mi] = part;
  }
  __syncthreads();
  if (tid < H) s_read[tid] = fmaxf(s_pp[0][tid] + s_pp[1][tid], 0.f);
  __syncthreads();
  if (tid < VOC) {
    float o = bo[tid];
#pragma unroll
    for (int i = 0; i < H; ++i) o = fmaf(s_read[i], wo[i * VOC + tid], o);
    out[b * VOC + tid] = o;
  }
}

// ---------------------------------------------------------------------------
extern "C" void kernel_launch(void* const* d_in, const int* in_sizes, int n_in,
                              void* d_out, int out_size, void* d_ws, size_t ws_size,
                              hipStream_t stream) {
  const int*   x     = (const int*)d_in[0];
  const float* embed = (const float*)d_in[1];
  const float* w1    = (const float*)d_in[2];
  const float* b1    = (const float*)d_in[3];
  const float* w2    = (const float*)d_in[4];
  const float* b2    = (const float*)d_in[5];
  const float* ln_g  = (const float*)d_in[6];
  const float* ln_b  = (const float*)d_in[7];
  const float* wk    = (const float*)d_in[8];
  const float* wv    = (const float*)d_in[9];
  const float* wq    = (const float*)d_in[10];
  const float* wo    = (const float*)d_in[11];
  const float* bo    = (const float*)d_in[12];
  float* out = (float*)d_out;

  bf16* ws = (bf16*)d_ws;
  const int total = 2 * H2 * H + 2 * H * H;   // 24576 bf16 elements (~48 KB)
  prep_weights_kernel<<<(total + 255) / 256, 256, 0, stream>>>(w1, w2, wk, wv, ws);

  const bf16* w1T = ws;
  const bf16* w2T = w1T + H2 * H;
  const bf16* wkT = w2T + H * H2;
  const bf16* wvT = wkT + H * H;

  const int B = in_sizes[0] / SEQL;           // 256
  delta_fused_kernel<<<B, 128, 0, stream>>>(x, embed, b1, b2, ln_g, ln_b,
                                            wq, wo, bo, w1T, w2T, wkT, wvT, out);
}
